// EdgeAwareBlur_76029511074372
// MI455X (gfx1250) — compile-verified, matrix operand order checked by reading, not run on hardware
//
#include <hip/hip_runtime.h>
#include <hip/hip_bf16.h>

// EdgeAwareBlur for MI455X (gfx1250, wave32).
// Pipeline (all on `stream`):
//   K0: init per-batch min/max slots
//   K1: Sobel energy + channel-mean via v_wmma_f32_16x16x32_f16 (A=ones trick),
//       block min/max reduction + uint-ordered atomics (edge >= 0)
//   K2: per-pixel 7-tap Gaussian weights, normalized, padded to 8 floats
//   K3: fused separable blur (H pass into LDS, V pass to output) -- avoids a
//       64MB intermediate (saves 128MB of traffic vs unfused; x and g stay
//       L2-resident in the 192MB L2). Tile staging uses CDNA5 async
//       global->LDS loads (ASYNCcnt) so no VGPR round-trip is needed.

typedef __attribute__((ext_vector_type(16))) _Float16 v16h;
typedef __attribute__((ext_vector_type(8)))  float    v8f;

#define BB 4
#define CC 64
#define HH 256
#define WW 256
#define HW (HH * WW)          // 65536
#define NPIX (BB * HW)        // 262144

// CDNA5 async global->LDS copy of one dword (per active lane).
// LDS dest VGPR carries the allocation-relative byte offset: for a flat
// shared-aperture pointer, addr[31:0] IS that offset (ISA 10.2 aperture rules).
__device__ __forceinline__ void async_copy_b32(const float* gsrc, void* lds_dst) {
  const unsigned int       ldsoff = (unsigned int)(uintptr_t)lds_dst;
  const unsigned long long gaddr  = (unsigned long long)(uintptr_t)gsrc;
  asm volatile("global_load_async_to_lds_b32 %0, %1, off"
               :: "v"(ldsoff), "v"(gaddr)
               : "memory");
}
__device__ __forceinline__ void wait_async_all() {
  asm volatile("s_wait_asynccnt 0" ::: "memory");
}

// ---------------------------------------------------------------- K0: init
__global__ void eab_init_mm(unsigned int* __restrict__ mm) {
  int t = threadIdx.x;
  if (t < 8) mm[t] = (t < 4) ? 0x7F800000u /*+inf*/ : 0u /*edge>=0 -> max init 0*/;
}

// ---------------------------------------------------------------- K1: edge
// One wave handles 16 consecutive pixels (always within one row: 256 % 16 == 0).
// Lane l supplies 16 f16 Sobel-energy values (pixel = l%16, channel half by l/16)
// as the B operand; A = ones, so D[:,n] = sum over 32 channels for pixel n
// (invariant to K-permutation within a column since A == 1 everywhere).
// Two accumulating WMMAs cover 64 channels. edge = sum / 64.
__global__ void __launch_bounds__(256)
eab_edge(const float* __restrict__ x, float* __restrict__ edge,
         unsigned int* __restrict__ mm) {
  const int tid  = threadIdx.x;
  const int lane = tid & 31;
  const int wib  = tid >> 5;                          // wave in block (0..7)
  const int base = (blockIdx.x * 8 + wib) * 16;       // first pixel of this wave
  const int col  = lane & 15;
  const int p    = base + col;                        // this lane's pixel
  const int b    = p >> 16;
  const int h    = (p >> 8) & 255;
  const int w    = p & 255;
  const float* xb = x + (b * CC << 16);

  v16h a;
#pragma unroll
  for (int i = 0; i < 16; ++i) a[i] = (_Float16)1.0f;
  v8f acc;
#pragma unroll
  for (int i = 0; i < 8; ++i) acc[i] = 0.0f;

  const int hm = h - 1, hp2 = h + 1, wm = w - 1, wp = w + 1;
  const int hmc = max(hm, 0), hpc = min(hp2, 255);
  const int wmc = max(wm, 0), wpc = min(wp, 255);
  const float mh0 = (hm >= 0) ? 1.f : 0.f, mh2 = (hp2 <= 255) ? 1.f : 0.f;
  const float mw0 = (wm >= 0) ? 1.f : 0.f, mw2 = (wp <= 255) ? 1.f : 0.f;
  const int r0 = (hmc << 8), r1 = (h << 8), r2 = (hpc << 8);
  const int chHalf = (lane < 16) ? 0 : 16;

#pragma unroll
  for (int iter = 0; iter < 2; ++iter) {
    v16h bm;
    const int ch0 = iter * 32 + chHalf;
#pragma unroll
    for (int j = 0; j < 16; ++j) {
      const float* xc = xb + ((ch0 + j) << 16);
      // 3x3 neighborhood (center unused), zero-padded via clamp+mask
      float x00 = xc[r0 + wmc] * (mh0 * mw0);
      float x01 = xc[r0 + w]   *  mh0;
      float x02 = xc[r0 + wpc] * (mh0 * mw2);
      float x10 = xc[r1 + wmc] *  mw0;
      float x12 = xc[r1 + wpc] *  mw2;
      float x20 = xc[r2 + wmc] * (mh2 * mw0);
      float x21 = xc[r2 + w]   *  mh2;
      float x22 = xc[r2 + wpc] * (mh2 * mw2);
      float gx = (x02 - x00) + 2.0f * (x12 - x10) + (x22 - x20);
      float gy = (x20 - x00) + 2.0f * (x21 - x01) + (x22 - x02);
      bm[j] = (_Float16)(gx * gx + gy * gy);
    }
    acc = __builtin_amdgcn_wmma_f32_16x16x32_f16(
        false, a, false, bm, (short)0, acc, false, false);
  }

  // D layout: VGPR0 holds row M=0 (lanes 0-15, N=lane) and M=8 (lanes 16-31,
  // N=lane-16); all rows equal (A=ones), so every lane has its column's sum.
  const float val = acc[0] * (1.0f / 64.0f);
  if (lane < 16) edge[p] = val;

  // Block min/max (128 distinct pixels; lanes>=16 hold duplicate column values,
  // which do not affect min/max). Block spans exactly one batch (65536%128==0).
  __shared__ float smin[256], smax[256];
  smin[tid] = val; smax[tid] = val;
  __syncthreads();
#pragma unroll
  for (int s = 128; s > 0; s >>= 1) {
    if (tid < s) {
      smin[tid] = fminf(smin[tid], smin[tid + s]);
      smax[tid] = fmaxf(smax[tid], smax[tid + s]);
    }
    __syncthreads();
  }
  if (tid == 0) {
    atomicMin(&mm[b],     __float_as_uint(smin[0]));   // valid: edge >= 0
    atomicMax(&mm[4 + b], __float_as_uint(smax[0]));
  }
}

// ---------------------------------------------------------------- K2: weights
__global__ void __launch_bounds__(256)
eab_gauss(const float* __restrict__ edge, const unsigned int* __restrict__ mm,
          float* __restrict__ g) {
  const int p = blockIdx.x * 256 + threadIdx.x;
  const int b = p >> 16;
  const float emin = __uint_as_float(mm[b]);
  const float emax = __uint_as_float(mm[4 + b]);
  const float e = (edge[p] - emin) / (emax + 1e-6f);   // faithful: denom = emax+eps
  const float sigma = 0.6f + 0.6f * e;
  const float inv2s2 = 1.0f / (2.0f * sigma * sigma);
  float wk[7]; float s = 0.0f;
#pragma unroll
  for (int k = 0; k < 7; ++k) {
    const float off = (float)(k - 3);
    wk[k] = __expf(-(off * off) * inv2s2);
    s += wk[k];
  }
  const float inv = 1.0f / s;
  float4 g0 = make_float4(wk[0]*inv, wk[1]*inv, wk[2]*inv, wk[3]*inv);
  float4 g1 = make_float4(wk[4]*inv, wk[5]*inv, wk[6]*inv, 0.0f);
  float4* gp = (float4*)(g + ((long long)p << 3));
  gp[0] = g0; gp[1] = g1;
}

// ---------------------------------------------------------------- K3: fused blur
#define TH 16
#define TW 64
__global__ void __launch_bounds__(256)
eab_blur(const float* __restrict__ x, const float* __restrict__ g,
         float* __restrict__ out) {
  // grid: (64 tiles [4 w-tiles x 16 h-tiles], C=64, B=4)
  const int wt = blockIdx.x & 3;
  const int ht = blockIdx.x >> 2;
  const int c  = blockIdx.y;
  const int b  = blockIdx.z;
  const int h0 = ht * TH, w0 = wt * TW;
  const int tid = threadIdx.x;

  __shared__ float xs[TH + 6][TW + 8];   // 22 x 72 (stride pad)
  __shared__ float hb[TH + 6][TW];       // horizontally blurred rows

  const float* xc = x + (((b << 6) + c) << 16);
  const float* gb = g + ((long long)(b << 16) << 3);
  const int obase = ((b << 6) + c) << 16;

  // Stage x tile (rows h0-3..h0+18, cols w0-3..w0+66) with CDNA5 async
  // global->LDS loads (clamped addresses; zero-pad fixed up below).
  for (int idx = tid; idx < 22 * 70; idx += 256) {
    const int r = idx / 70, cc2 = idx % 70;
    const int gh = h0 - 3 + r, gw = w0 - 3 + cc2;
    const int ghc = min(max(gh, 0), 255), gwc = min(max(gw, 0), 255);
    async_copy_b32(xc + (ghc << 8) + gwc, &xs[r][cc2]);
  }
  wait_async_all();
  __syncthreads();

  // Zero-pad fix-up: only boundary tiles have OOB halo cells.
  if (h0 == 0 || h0 == (HH - TH) || wt == 0 || wt == 3) {
    for (int idx = tid; idx < 22 * 70; idx += 256) {
      const int r = idx / 70, cc2 = idx % 70;
      const int gh = h0 - 3 + r, gw = w0 - 3 + cc2;
      if ((unsigned)gh >= 256u || (unsigned)gw >= 256u) xs[r][cc2] = 0.0f;
    }
    __syncthreads();
  }

  // Horizontal pass: hb[r][w] uses g at (gh, w0+w); out-of-image rows -> 0
  // (they are the zero-pad for the vertical pass).
  for (int idx = tid; idx < 22 * TW; idx += 256) {
    const int r = idx >> 6, w = idx & 63;
    const int gh = h0 - 3 + r;
    float v = 0.0f;
    if ((unsigned)gh < 256u) {
      const float4* gp = (const float4*)(gb + (((gh << 8) + (w0 + w)) << 3));
      const float4 g0 = gp[0], g1 = gp[1];
      const float wk[7] = {g0.x, g0.y, g0.z, g0.w, g1.x, g1.y, g1.z};
#pragma unroll
      for (int k = 0; k < 7; ++k) v += xs[r][w + k] * wk[k];
    }
    hb[r][w] = v;
  }
  __syncthreads();

  // Vertical pass: out[h0+r][w0+w] = sum_k hb[r+k][w] * g[h0+r][w0+w][k]
  for (int idx = tid; idx < TH * TW; idx += 256) {
    const int r = idx >> 6, w = idx & 63;
    const int gh = h0 + r;
    const float4* gp = (const float4*)(gb + (((gh << 8) + (w0 + w)) << 3));
    const float4 g0 = gp[0], g1 = gp[1];
    const float wk[7] = {g0.x, g0.y, g0.z, g0.w, g1.x, g1.y, g1.z};
    float v = 0.0f;
#pragma unroll
    for (int k = 0; k < 7; ++k) v += hb[r + k][w] * wk[k];
    out[obase + (gh << 8) + (w0 + w)] = v;
  }
}

// ---------------------------------------------------------------- launch
extern "C" void kernel_launch(void* const* d_in, const int* in_sizes, int n_in,
                              void* d_out, int out_size, void* d_ws, size_t ws_size,
                              hipStream_t stream) {
  (void)in_sizes; (void)n_in; (void)out_size; (void)ws_size;
  const float* x = (const float*)d_in[0];
  float* out = (float*)d_out;
  char* ws = (char*)d_ws;

  float*        edge = (float*)ws;                                  // 1 MB
  unsigned int* mm   = (unsigned int*)(ws + (size_t)NPIX * 4);      // 32 B
  float*        g    = (float*)(ws + (size_t)NPIX * 4 + 256);       // 8 MB

  eab_init_mm<<<1, 8, 0, stream>>>(mm);
  eab_edge<<<NPIX / 128, 256, 0, stream>>>(x, edge, mm);
  eab_gauss<<<NPIX / 256, 256, 0, stream>>>(edge, mm, g);
  dim3 grid(64, CC, BB);
  eab_blur<<<grid, 256, 0, stream>>>(x, g, out);
}
